// VoxelSDF_19215683682576
// MI455X (gfx1250) — matrix operand order repaired and checked by reading, not compiled, not run on hardware
//
#include <hip/hip_runtime.h>

#define G1     33
#define VOX    (G1*G1*G1)      // 35937
#define LATENT 32
#define HID    128
#define NPTS   1048576

typedef __attribute__((ext_vector_type(16))) _Float16 v16h;
typedef __attribute__((ext_vector_type(8)))  _Float16 v8h;
typedef __attribute__((ext_vector_type(8)))  float    v8f;

// d_ws layout: [0, EMB_H_BYTES) = repacked f16 emb [voxel][32]
//              [FRAG_OFF, +40*512*2) = f16 WMMA B-fragments (8 w1 tiles + 32 w2 tiles)
#define EMB_H_BYTES (VOX*LATENT*2)   // 2,299,968 (divisible by 64 -> aligned)
#define FRAG_OFF    EMB_H_BYTES
#define NTILES      40

// ---------- pre-pass 1: transpose emb from [c][voxel] to [voxel][c], f32 -> f16 ----------
__global__ void repack_emb_kernel(const float* __restrict__ emb, _Float16* __restrict__ out) {
  int i = blockIdx.x * 256 + threadIdx.x;          // i = v*32 + c
  if (i >= VOX * LATENT) return;
  int v = i >> 5, c = i & 31;
  out[i] = (_Float16)emb[c * VOX + v];
}

// ---------- pre-pass 2: build f16 B-fragments in WMMA lane order ----------
// B order used here (contiguous K, matches A): lane = (K>=16)*16 + col, halves j -> K = (lane>>4)*16 + j
__global__ void build_frags_kernel(const float* __restrict__ w1, const float* __restrict__ w2,
                                   _Float16* __restrict__ frag) {
  int t = blockIdx.x * 256 + threadIdx.x;          // t = (tile*32 + lane)*16 + j
  if (t >= NTILES * 512) return;
  int tile = t >> 9, lane = (t >> 4) & 31, j = t & 15;
  int col16 = lane & 15, hi = lane >> 4;
  float val;
  if (tile < 8) {                                  // w1 [32,128], tile nt = tile, K block = 0..31
    int col = tile * 16 + col16;
    int K   = hi * 16 + j;
    val = w1[K * HID + col];
  } else {                                         // w2 [128,128], tiles (nt, kb)
    int tt = tile - 8, nt = tt >> 2, kb = tt & 3;
    int col = nt * 16 + col16;
    int K   = kb * 32 + hi * 16 + j;
    val = w2[K * HID + col];
  }
  frag[t] = (_Float16)val;
}

// ---------- main fused kernel: trilinear gather + 3-layer MLP via WMMA ----------
__global__ void __launch_bounds__(256) voxel_sdf_kernel(
    const float* __restrict__ points,
    const _Float16* __restrict__ embh,
    const _Float16* __restrict__ frags,
    const float* __restrict__ b1,
    const float* __restrict__ b2,
    const float* __restrict__ w3,
    const float* __restrict__ b3,
    float* __restrict__ out)
{
  __shared__ _Float16 h1[8][16][HID];              // 32 KB: per-wave 16x128 f16 hidden tile
  const int lane = threadIdx.x & 31;
  const int w    = threadIdx.x >> 5;               // wave in block (0..7)
  const int m    = lane & 15;                      // point row within tile
  const int hi   = lane >> 4;                      // lane half (A/B K-half select)
  const int p0   = blockIdx.x * 128 + w * 16;
  const int pt   = p0 + m;

  // ---- trilinear sample: each lane computes the 16 channels of its A-fragment slot ----
  float px = points[pt*3+0], py = points[pt*3+1], pz = points[pt*3+2];
  float cx = (px + 1.f) * 16.f, cy = (py + 1.f) * 16.f, cz = (pz + 1.f) * 16.f;
  float fxx = floorf(cx), fyy = floorf(cy), fzz = floorf(cz);
  int x0 = (int)fxx, y0 = (int)fyy, z0 = (int)fzz;
  float fx = cx - fxx, fy = cy - fyy, fz = cz - fzz;
  float wxc[2] = {1.f - fx, fx}, wyc[2] = {1.f - fy, fy}, wzc[2] = {1.f - fz, fz};

  float acc[16];
  #pragma unroll
  for (int j = 0; j < 16; ++j) acc[j] = 0.f;

  #pragma unroll
  for (int c = 0; c < 8; ++c) {
    int xi = x0 + (c & 1), yi = y0 + ((c >> 1) & 1), zi = z0 + (c >> 2);
    float wgt = wzc[c >> 2] * wyc[(c >> 1) & 1] * wxc[c & 1];
    bool valid = ((unsigned)xi < (unsigned)G1) & ((unsigned)yi < (unsigned)G1) &
                 ((unsigned)zi < (unsigned)G1);
    wgt = valid ? wgt : 0.f;
    int xc = min(max(xi, 0), G1-1), yc = min(max(yi, 0), G1-1), zc = min(max(zi, 0), G1-1);
    int v = (zc * G1 + yc) * G1 + xc;
    const _Float16* p = embh + v * LATENT + hi * 16; // lane<16: ch 0..15; lane>=16: 16..31
    v8h lo = *(const v8h*)(p);
    v8h hh = *(const v8h*)(p + 8);
    #pragma unroll
    for (int j = 0; j < 8; ++j) acc[j]     += wgt * (float)lo[j];
    #pragma unroll
    for (int j = 0; j < 8; ++j) acc[8 + j] += wgt * (float)hh[j];
  }
  v16h a;
  #pragma unroll
  for (int j = 0; j < 16; ++j) a[j] = (_Float16)acc[j];

  // ---- layer 1: [16,32] @ [32,128] via 8 WMMAs (bias preloaded in C), ReLU -> LDS f16 ----
  #pragma unroll
  for (int nt = 0; nt < 8; ++nt) {
    v16h bf = *(const v16h*)(frags + (nt * 32 + lane) * 16);
    float bb = b1[nt * 16 + m];
    v8f c;
    #pragma unroll
    for (int r = 0; r < 8; ++r) c[r] = bb;
    v8f d = __builtin_amdgcn_wmma_f32_16x16x32_f16(false, a, false, bf, (short)0, c, false, false);
    #pragma unroll
    for (int r = 0; r < 8; ++r)
      h1[w][hi * 8 + r][nt * 16 + m] = (_Float16)fmaxf(d[r], 0.f);  // D: row = r + hi*8, col = m
  }
  __syncthreads();

  // ---- rebuild A-fragments for layer 2 (K = 128 -> 4 chunks of 32) from LDS ----
  v16h a2[4];
  #pragma unroll
  for (int kb = 0; kb < 4; ++kb) {
    #pragma unroll
    for (int jj = 0; jj < 8; ++jj) {
      int j  = jj * 2;
      int kl = j + hi * 16;      // A-layout local K for halves (j, j+1)
      const _Float16* sp = &h1[w][m][kb * 32 + kl];
      a2[kb][j]     = sp[0];
      a2[kb][j + 1] = sp[1];
    }
  }

  // ---- layer 2: [16,128] @ [128,128] via 8x4 chained WMMAs; fuse layer-3 dot on D frags ----
  float partial[8];
  #pragma unroll
  for (int r = 0; r < 8; ++r) partial[r] = 0.f;

  #pragma unroll
  for (int nt = 0; nt < 8; ++nt) {
    float bb = b2[nt * 16 + m];
    v8f c;
    #pragma unroll
    for (int r = 0; r < 8; ++r) c[r] = bb;
    #pragma unroll
    for (int kb = 0; kb < 4; ++kb) {
      v16h bf = *(const v16h*)(frags + ((8 + nt * 4 + kb) * 32 + lane) * 16);
      c = __builtin_amdgcn_wmma_f32_16x16x32_f16(false, a2[kb], false, bf, (short)0, c, false, false);
    }
    float w3v = w3[nt * 16 + m];                    // channel nt*16 + col held by this lane
    #pragma unroll
    for (int r = 0; r < 8; ++r) partial[r] += fmaxf(c[r], 0.f) * w3v;
  }

  // ---- layer 3 finish: reduce 128-ch dot across the 16 lanes sharing each row-half ----
  #pragma unroll
  for (int mask = 1; mask < 16; mask <<= 1) {
    #pragma unroll
    for (int r = 0; r < 8; ++r) partial[r] += __shfl_xor(partial[r], mask, 32);
  }
  if (m == 0) {
    float bb3 = b3[0];
    #pragma unroll
    for (int r = 0; r < 8; ++r)
      out[p0 + hi * 8 + r] = tanhf(partial[r] + bb3);  // lane0: rows 0..7, lane16: rows 8..15
  }
}

extern "C" void kernel_launch(void* const* d_in, const int* in_sizes, int n_in,
                              void* d_out, int out_size, void* d_ws, size_t ws_size,
                              hipStream_t stream) {
  const float* points = (const float*)d_in[0];
  const float* emb    = (const float*)d_in[1];
  const float* w1     = (const float*)d_in[2];
  const float* b1     = (const float*)d_in[3];
  const float* w2     = (const float*)d_in[4];
  const float* b2     = (const float*)d_in[5];
  const float* w3     = (const float*)d_in[6];
  const float* b3     = (const float*)d_in[7];
  float* out = (float*)d_out;

  _Float16* embh  = (_Float16*)d_ws;
  _Float16* frags = (_Float16*)((char*)d_ws + FRAG_OFF);

  repack_emb_kernel<<<(VOX * LATENT + 255) / 256, 256, 0, stream>>>(emb, embh);
  build_frags_kernel<<<(NTILES * 512 + 255) / 256, 256, 0, stream>>>(w1, w2, frags);
  voxel_sdf_kernel<<<NPTS / 128, 256, 0, stream>>>(points, embh, frags, b1, b2, w3, b3, out);
}
